// DiffMambaLayer_76184129896655
// MI455X (gfx1250) — matrix operand order repaired and checked by reading, not compiled, not run on hardware
//
#include <hip/hip_runtime.h>
#include <math.h>

// ---------------------------------------------------------------------------
// DiffMamba layer for MI455X (gfx1250, wave32).
//   B=4, DIM=384, L=3136, M = B*L = 12544, N_state=8, DCONV=4, DTR=24.
// GEMMs run on v_wmma_f32_16x16x32_bf16 (fp32 accum), 32x64 output tile per
// wave (8 accumulators, 2 A-frags x 4 B-frags per K-step, K fully unrolled).
// Scan is a 3-phase chunked parallel scan (64 chunks x 49 steps). LN kernels
// stage 384x32 tiles in LDS for the (B,C,L) <-> (B,L,C) transposes.
// ---------------------------------------------------------------------------

typedef unsigned short u16;
typedef __attribute__((ext_vector_type(16))) __bf16 v16bf;
typedef __attribute__((ext_vector_type(8)))  float  v8f;

#define DIMC   384
#define LSEQ   3136
#define NB     4
#define MTOK   (NB * LSEQ)     // 12544
#define NSTATE 8
#define NCHUNK 64
#define CHL    49              // 64 * 49 == 3136

__device__ __forceinline__ u16 f2bf(float f) {
  union { float f; unsigned u; } x; x.f = f;
  unsigned r = x.u + 0x7fffu + ((x.u >> 16) & 1u);   // round-to-nearest-even
  return (u16)(r >> 16);
}

__device__ __forceinline__ float wred(float v) {
  #pragma unroll
  for (int o = 16; o > 0; o >>= 1) v += __shfl_xor(v, o, 32);
  return v;
}

// ---------------------------------------------------------------------------
// Weight convert fp32 -> bf16 with zero padding (rs x cs) -> (rd x cd).
// ---------------------------------------------------------------------------
__global__ void pad_convert_kernel(const float* __restrict__ src, u16* __restrict__ dst,
                                   int rs, int cs, int rd, int cd) {
  int g = blockIdx.x * blockDim.x + threadIdx.x;
  if (g >= rd * cd) return;
  int r = g / cd, c = g % cd;
  float v = (r < rs && c < cs) ? src[r * cs + c] : 0.f;
  dst[g] = f2bf(v);
}

// ---------------------------------------------------------------------------
// LayerNorm1 + transpose: x (B,C,L) -> xn_bf16 (B*L, C) and xf (B*L, C) f32.
// One block handles 32 consecutive l of one batch via a 384x32 LDS tile.
// ---------------------------------------------------------------------------
__global__ void ln_transpose_kernel(const float* __restrict__ x, const float* __restrict__ w,
                                    const float* __restrict__ b, u16* __restrict__ xn_bf,
                                    float* __restrict__ xf) {
  __shared__ float tile[DIMC * 33];
  int blk = blockIdx.x;
  int bb  = blk / (LSEQ / 32);
  int l0  = (blk % (LSEQ / 32)) * 32;
  int t = threadIdx.x;
  #pragma unroll 4
  for (int it = 0; it < 48; ++it) {             // 384*32 / 256
    int lin = t + it * 256;
    int c = lin >> 5, dl = lin & 31;
    tile[c * 33 + dl] = x[((size_t)bb * DIMC + c) * LSEQ + l0 + dl];
  }
  __syncthreads();
  int wave = t >> 5, lane = t & 31;
  for (int st = 0; st < 4; ++st) {
    int dl = wave * 4 + st;
    size_t tok = (size_t)bb * LSEQ + l0 + dl;
    float vv[12], s1 = 0.f, s2 = 0.f;
    #pragma unroll
    for (int j = 0; j < 12; ++j) {
      float v = tile[(lane + 32 * j) * 33 + dl];
      vv[j] = v; s1 += v; s2 += v * v;
    }
    s1 = wred(s1); s2 = wred(s2);
    float mean = s1 * (1.f / DIMC);
    float var  = s2 * (1.f / DIMC) - mean * mean;
    float inv  = rsqrtf(var + 1e-5f);
    #pragma unroll
    for (int j = 0; j < 12; ++j) {
      int c = lane + 32 * j;
      xf[tok * DIMC + c] = vv[j];
      xn_bf[tok * DIMC + c] = f2bf((vv[j] - mean) * inv * w[c] + b[c]);
    }
  }
}

// ---------------------------------------------------------------------------
// bf16 WMMA fragment union: per-lane K-runs of A and B rows are contiguous
// 16B slabs in row-major bf16, so each fragment is exactly 2x b128 loads.
// ---------------------------------------------------------------------------
union FragBF { uint4 u[2]; v16bf v; };

// ---------------------------------------------------------------------------
// Multi-tile bf16 WMMA GEMM: C(MxN) = A(MxK) @ W(NxK)^T, fp32 out.
// One wave owns a 32x64 tile: 2 M-subtiles x 4 N-subtiles = 8 accumulators.
// Per K-step: 2 A-frags + 4 B-frags (12x b128) feed 8 WMMAs (~21 flops/byte
// of fragment traffic). KC compile-time -> full unroll, loads hoisted across
// earlier WMMAs by the scheduler. Requires M%32==0, N%64==0, K==KC.
// EPI==1: softplus(x + bias[n])  (dt_proj epilogue).
// ---------------------------------------------------------------------------
template <int EPI, int KC>
__global__ void gemm_bf16_mt_kernel(const u16* __restrict__ A, const u16* __restrict__ W,
                                    const float* __restrict__ bias, float* __restrict__ C,
                                    int M, int N) {
  int gwave = (blockIdx.x * blockDim.x + threadIdx.x) >> 5;
  int lane  = threadIdx.x & 31;
  int NT = N >> 6;
  int tiles = (M >> 5) * NT;
  if (gwave >= tiles) return;
  int tm = gwave / NT, tn = gwave % NT;
  int m = lane & 15, h = lane >> 4;

  const u16* ar[2];
  ar[0] = A + (size_t)(tm * 32 + m) * KC;
  ar[1] = ar[0] + (size_t)16 * KC;
  const u16* wr[4];
  #pragma unroll
  for (int j = 0; j < 4; ++j) wr[j] = W + (size_t)(tn * 64 + j * 16 + m) * KC;

  v8f zero = {0.f, 0.f, 0.f, 0.f, 0.f, 0.f, 0.f, 0.f};
  v8f acc[2][4];
  #pragma unroll
  for (int s = 0; s < 2; ++s)
    #pragma unroll
    for (int j = 0; j < 4; ++j) acc[s][j] = zero;

  #pragma unroll
  for (int k0 = 0; k0 < KC; k0 += 32) {
    if (k0 + 32 < KC) {
      __builtin_prefetch((const void*)(ar[0] + k0 + 32), 0, 3);   // global_prefetch_b8
      __builtin_prefetch((const void*)(wr[0] + k0 + 32), 0, 3);
    }
    FragBF fa[2], fb[4];
    #pragma unroll
    for (int s = 0; s < 2; ++s) {
      const uint4* pa = (const uint4*)(ar[s] + k0);
      fa[s].u[0] = pa[h];          // k = h*8    .. h*8+7
      fa[s].u[1] = pa[2 + h];      // k = 16+h*8 ..
    }
    #pragma unroll
    for (int j = 0; j < 4; ++j) {
      const uint4* pb = (const uint4*)(wr[j] + k0);
      fb[j].u[0] = pb[h];          // k = h*8    .. h*8+7
      fb[j].u[1] = pb[2 + h];      // k = 16+h*8 ..
    }
    #pragma unroll
    for (int s = 0; s < 2; ++s)
      #pragma unroll
      for (int j = 0; j < 4; ++j)
        acc[s][j] = __builtin_amdgcn_wmma_f32_16x16x32_bf16(false, fa[s].v, false, fb[j].v,
                                                            (short)0, acc[s][j], false, false);
  }

  #pragma unroll
  for (int s = 0; s < 2; ++s) {
    #pragma unroll
    for (int j = 0; j < 4; ++j) {
      int col = tn * 64 + j * 16 + m;
      float bcol = (EPI == 1) ? bias[col] : 0.f;
      #pragma unroll
      for (int r = 0; r < 8; ++r) {
        float v = acc[s][j][r];
        if (EPI == 1) {
          v += bcol;
          v = (v > 20.f) ? v : log1pf(__expf(v));        // softplus
        }
        C[(size_t)(tm * 32 + s * 16 + h * 8 + r) * N + col] = v;
      }
    }
  }
}

// ---------------------------------------------------------------------------
// Single-tile bf16 WMMA GEMM (for N not divisible by 64, i.e. x_proj N=48).
// ---------------------------------------------------------------------------
__global__ void gemm_bf16_kernel(const u16* __restrict__ A, const u16* __restrict__ W,
                                 float* __restrict__ C, int M, int N, int K) {
  int gwave = (blockIdx.x * blockDim.x + threadIdx.x) >> 5;
  int lane  = threadIdx.x & 31;
  int NT = N >> 4;
  int tiles = (M >> 4) * NT;
  if (gwave >= tiles) return;
  int tm = gwave / NT, tn = gwave % NT;
  int m = lane & 15, h = lane >> 4;

  const u16* arow = A + (size_t)(tm * 16 + m) * K;
  const u16* wrow = W + (size_t)(tn * 16 + m) * K;
  v8f acc = {0.f, 0.f, 0.f, 0.f, 0.f, 0.f, 0.f, 0.f};

  for (int k0 = 0; k0 < K; k0 += 32) {
    if (k0 + 32 < K) {
      __builtin_prefetch((const void*)(arow + k0 + 32), 0, 3);
      __builtin_prefetch((const void*)(wrow + k0 + 32), 0, 3);
    }
    FragBF fa, fb;
    const uint4* pa = (const uint4*)(arow + k0);
    fa.u[0] = pa[h];
    fa.u[1] = pa[2 + h];
    const uint4* pb = (const uint4*)(wrow + k0);
    fb.u[0] = pb[h];
    fb.u[1] = pb[2 + h];
    acc = __builtin_amdgcn_wmma_f32_16x16x32_bf16(false, fa.v, false, fb.v,
                                                  (short)0, acc, false, false);
  }

  int col = tn * 16 + m;
  #pragma unroll
  for (int r = 0; r < 8; ++r)
    C[(size_t)(tm * 16 + h * 8 + r) * N + col] = acc[r];
}

// ---------------------------------------------------------------------------
// Depthwise causal conv (DCONV=4) + bias + SiLU.  xz rows are (xs[384]|z[384]).
// ---------------------------------------------------------------------------
__global__ void conv_silu_kernel(const float* __restrict__ xz, const float* __restrict__ cw,
                                 const float* __restrict__ cb, float* __restrict__ xs,
                                 u16* __restrict__ xsb) {
  size_t g = (size_t)blockIdx.x * 256 + threadIdx.x;
  if (g >= (size_t)MTOK * DIMC) return;
  int d = (int)(g % DIMC);
  size_t tok = g / DIMC;
  int l = (int)(tok % LSEQ);
  size_t brow = tok - l;               // b*L
  float acc = cb[d];
  #pragma unroll
  for (int j = 0; j < 4; ++j) {
    int ll = l - 3 + j;
    if (ll >= 0) acc += cw[d * 4 + j] * xz[(brow + ll) * 768 + d];
  }
  float sv = acc / (1.f + __expf(-acc));
  xs[g] = sv;
  xsb[g] = f2bf(sv);
}

// ---------------------------------------------------------------------------
// Pack dt input: x_dbl[:, :24] -> bf16 (M x 32), cols 24..31 zero.
// ---------------------------------------------------------------------------
__global__ void pack_dt_kernel(const float* __restrict__ xdbl, u16* __restrict__ dtin) {
  size_t g = (size_t)blockIdx.x * 256 + threadIdx.x;
  if (g >= (size_t)MTOK * 32) return;
  int c = (int)(g & 31);
  size_t tok = g >> 5;
  float v = (c < 24) ? xdbl[tok * 48 + c] : 0.f;
  dtin[g] = f2bf(v);
}

// ---------------------------------------------------------------------------
// Scan phase 1: per (b,d,chunk) compute prod(dA) and h_end with h_in = 0.
// Consecutive threads = consecutive d -> coalesced dt/xs reads.
// ---------------------------------------------------------------------------
__global__ void scan_phase1_kernel(const float* __restrict__ dt, const float* __restrict__ xs,
                                   const float* __restrict__ xdbl, const float* __restrict__ alog,
                                   float* __restrict__ cs) {
  int g = blockIdx.x * 256 + threadIdx.x;
  if (g >= NB * DIMC * NCHUNK) return;
  int d  = g % DIMC;
  int bb = (g / DIMC) & 3;
  int ch = g / (NB * DIMC);
  float a[NSTATE], ap[NSTATE], h[NSTATE];
  #pragma unroll
  for (int n = 0; n < NSTATE; ++n) {
    a[n] = -__expf(alog[d * NSTATE + n]); ap[n] = 1.f; h[n] = 0.f;
  }
  int l0 = ch * CHL;
  for (int st = 0; st < CHL; ++st) {
    size_t tok = (size_t)bb * LSEQ + l0 + st;
    float dtv = dt[tok * DIMC + d];
    float dx  = dtv * xs[tok * DIMC + d];
    const float* bc = xdbl + tok * 48;
    #pragma unroll
    for (int n = 0; n < NSTATE; ++n) {
      float da = __expf(dtv * a[n]);
      ap[n] *= da;
      h[n] = da * h[n] + dx * bc[24 + n];
    }
  }
  size_t base = ((size_t)(bb * DIMC + d) * NCHUNK + ch) * 16;
  #pragma unroll
  for (int n = 0; n < NSTATE; ++n) { cs[base + n] = ap[n]; cs[base + 8 + n] = h[n]; }
}

// ---------------------------------------------------------------------------
// Scan phase 2: serial chunk fixup per (b,d,n): h_in for every chunk.
// ---------------------------------------------------------------------------
__global__ void scan_fix_kernel(const float* __restrict__ cs, float* __restrict__ hin) {
  int g = blockIdx.x * 256 + threadIdx.x;
  if (g >= NB * DIMC * NSTATE) return;
  int n  = g & 7;
  int bd = g >> 3;
  float h = 0.f;
  for (int ch = 0; ch < NCHUNK; ++ch) {
    size_t base = (size_t)bd * NCHUNK + ch;
    hin[base * 8 + n] = h;
    h = cs[base * 16 + n] * h + cs[base * 16 + 8 + n];
  }
}

// ---------------------------------------------------------------------------
// Scan phase 3: replay chunks with correct h_in, fuse y = sum h*C, D-skip,
// SiLU(z) gate; emit bf16 for out_proj WMMA.
// ---------------------------------------------------------------------------
__global__ void scan_phase3_kernel(const float* __restrict__ dt, const float* __restrict__ xs,
                                   const float* __restrict__ xdbl, const float* __restrict__ alog,
                                   const float* __restrict__ Dp, const float* __restrict__ xz,
                                   const float* __restrict__ hin, u16* __restrict__ gout) {
  int g = blockIdx.x * 256 + threadIdx.x;
  if (g >= NB * DIMC * NCHUNK) return;
  int d  = g % DIMC;
  int bb = (g / DIMC) & 3;
  int ch = g / (NB * DIMC);
  float a[NSTATE], h[NSTATE];
  size_t hb = ((size_t)(bb * DIMC + d) * NCHUNK + ch) * 8;
  #pragma unroll
  for (int n = 0; n < NSTATE; ++n) {
    a[n] = -__expf(alog[d * NSTATE + n]);
    h[n] = hin[hb + n];
  }
  float Dd = Dp[d];
  int l0 = ch * CHL;
  for (int st = 0; st < CHL; ++st) {
    size_t tok = (size_t)bb * LSEQ + l0 + st;
    float dtv = dt[tok * DIMC + d];
    float xv  = xs[tok * DIMC + d];
    float dx  = dtv * xv;
    const float* bc = xdbl + tok * 48;
    float y = 0.f;
    #pragma unroll
    for (int n = 0; n < NSTATE; ++n) {
      float da = __expf(dtv * a[n]);
      h[n] = da * h[n] + dx * bc[24 + n];
      y += h[n] * bc[32 + n];
    }
    float zv = xz[tok * 768 + DIMC + d];
    float gv = (y + xv * Dd) * (zv / (1.f + __expf(-zv)));
    gout[tok * DIMC + d] = f2bf(gv);
  }
}

// ---------------------------------------------------------------------------
// lam = sigmoid(sum(lambda_q))
// ---------------------------------------------------------------------------
__global__ void lambda_kernel(const float* __restrict__ lq, float* __restrict__ lam) {
  __shared__ float red[128];
  int t = threadIdx.x;
  float s = 0.f;
  for (int i = t; i < DIMC; i += 128) s += lq[i];
  red[t] = s; __syncthreads();
  for (int o = 64; o > 0; o >>= 1) { if (t < o) red[t] += red[t + o]; __syncthreads(); }
  if (t == 0) lam[0] = 1.f / (1.f + __expf(-red[0]));
}

// ---------------------------------------------------------------------------
// attn = LN(y1 - lam*y2); out = LN(xf + attn); transpose back to (B,C,L).
// ---------------------------------------------------------------------------
__global__ void final_kernel(const float* __restrict__ xf, const float* __restrict__ y1,
                             const float* __restrict__ y2, const float* __restrict__ lamp,
                             const float* __restrict__ sw, const float* __restrict__ sb,
                             const float* __restrict__ n2w, const float* __restrict__ n2b,
                             float* __restrict__ out) {
  __shared__ float tile[DIMC * 33];
  int blk = blockIdx.x;
  int bb  = blk / (LSEQ / 32);
  int l0  = (blk % (LSEQ / 32)) * 32;
  int t = threadIdx.x, wave = t >> 5, lane = t & 31;
  float lam = lamp[0];
  for (int st = 0; st < 4; ++st) {
    int dl = wave * 4 + st;
    size_t tok = (size_t)bb * LSEQ + l0 + dl;
    float vv[12], uu[12], s1 = 0.f, s2 = 0.f;
    #pragma unroll
    for (int j = 0; j < 12; ++j) {
      int c = lane + 32 * j;
      float v = y1[tok * DIMC + c] - lam * y2[tok * DIMC + c];
      vv[j] = v; s1 += v; s2 += v * v;
    }
    s1 = wred(s1); s2 = wred(s2);
    float mean = s1 * (1.f / DIMC);
    float inv  = rsqrtf(s2 * (1.f / DIMC) - mean * mean + 1e-5f);
    s1 = 0.f; s2 = 0.f;
    #pragma unroll
    for (int j = 0; j < 12; ++j) {
      int c = lane + 32 * j;
      float attn = (vv[j] - mean) * inv * sw[c] + sb[c];
      float u = xf[tok * DIMC + c] + attn;
      uu[j] = u; s1 += u; s2 += u * u;
    }
    s1 = wred(s1); s2 = wred(s2);
    float m2   = s1 * (1.f / DIMC);
    float inv2 = rsqrtf(s2 * (1.f / DIMC) - m2 * m2 + 1e-5f);
    #pragma unroll
    for (int j = 0; j < 12; ++j) {
      int c = lane + 32 * j;
      tile[c * 33 + dl] = (uu[j] - m2) * inv2 * n2w[c] + n2b[c];
    }
  }
  __syncthreads();
  #pragma unroll 4
  for (int it = 0; it < 48; ++it) {
    int lin = t + it * 256;
    int c = lin >> 5, dl = lin & 31;
    out[((size_t)bb * DIMC + c) * LSEQ + l0 + dl] = tile[c * 33 + dl];
  }
}

// ---------------------------------------------------------------------------
// Host orchestration.
// Input flattening assumed in setup_inputs() insertion order:
//   0: x
//   1..9 : m1 {in_proj_w, conv_w, conv_b, x_proj_w, dt_proj_w, dt_proj_b,
//              A_log, D, out_proj_w}
//   10..18: m2 (same order)
//   19..25: norm1_w, norm1_b, norm2_w, norm2_b, subln_w, subln_b, lambda_q
// ---------------------------------------------------------------------------
extern "C" void kernel_launch(void* const* d_in, const int* in_sizes, int n_in,
                              void* d_out, int out_size, void* d_ws, size_t ws_size,
                              hipStream_t stream) {
  (void)in_sizes; (void)n_in; (void)out_size; (void)ws_size;

  int i = 0;
  const float* x = (const float*)d_in[i++];
  const float* mw[2][9];
  for (int m = 0; m < 2; ++m)
    for (int j = 0; j < 9; ++j) mw[m][j] = (const float*)d_in[i++];
  const float* norm1_w = (const float*)d_in[i++];
  const float* norm1_b = (const float*)d_in[i++];
  const float* norm2_w = (const float*)d_in[i++];
  const float* norm2_b = (const float*)d_in[i++];
  const float* subln_w = (const float*)d_in[i++];
  const float* subln_b = (const float*)d_in[i++];
  const float* lambda_q = (const float*)d_in[i++];

  // Workspace sub-allocation (256B aligned). Total ~178 MB.
  char* base = (char*)d_ws;
  size_t off = 0;
  auto alloc = [&](size_t bytes) -> void* {
    void* p = base + off;
    off += (bytes + 255) & ~(size_t)255;
    return p;
  };
  u16*   xn_bf = (u16*)  alloc((size_t)MTOK * DIMC * 2);
  float* xf    = (float*)alloc((size_t)MTOK * DIMC * 4);
  float* ybuf0 = (float*)alloc((size_t)MTOK * DIMC * 4);
  float* ybuf1 = (float*)alloc((size_t)MTOK * DIMC * 4);
  float* xz    = (float*)alloc((size_t)MTOK * 768 * 4);
  float* xs    = (float*)alloc((size_t)MTOK * DIMC * 4);
  u16*   xsb   = (u16*)  alloc((size_t)MTOK * DIMC * 2);
  float* xdbl  = (float*)alloc((size_t)MTOK * 48 * 4);
  u16*   dtin  = (u16*)  alloc((size_t)MTOK * 32 * 2);
  float* dtb   = (float*)alloc((size_t)MTOK * DIMC * 4);
  float* cs    = (float*)alloc((size_t)NB * DIMC * NCHUNK * 16 * 4);
  float* hin   = (float*)alloc((size_t)NB * DIMC * NCHUNK * 8 * 4);
  u16*   gb    = (u16*)  alloc((size_t)MTOK * DIMC * 2);
  u16*   wb[2][4];
  for (int m = 0; m < 2; ++m) {
    wb[m][0] = (u16*)alloc((size_t)768 * 384 * 2);   // in_proj
    wb[m][1] = (u16*)alloc((size_t)48 * 384 * 2);    // x_proj (padded 40->48)
    wb[m][2] = (u16*)alloc((size_t)384 * 32 * 2);    // dt_proj (padded K 24->32)
    wb[m][3] = (u16*)alloc((size_t)384 * 384 * 2);   // out_proj
  }
  float* lam = (float*)alloc(4);
  float* ybuf[2] = {ybuf0, ybuf1};

  // Convert/pad weights to bf16.
  for (int m = 0; m < 2; ++m) {
    pad_convert_kernel<<<(768 * 384 + 255) / 256, 256, 0, stream>>>(mw[m][0], wb[m][0], 768, 384, 768, 384);
    pad_convert_kernel<<<(48 * 384 + 255) / 256, 256, 0, stream>>>(mw[m][3], wb[m][1], 40, 384, 48, 384);
    pad_convert_kernel<<<(384 * 32 + 255) / 256, 256, 0, stream>>>(mw[m][4], wb[m][2], 384, 24, 384, 32);
    pad_convert_kernel<<<(384 * 384 + 255) / 256, 256, 0, stream>>>(mw[m][8], wb[m][3], 384, 384, 384, 384);
  }

  // LN1 + transpose; lambda scalar.
  ln_transpose_kernel<<<NB * (LSEQ / 32), 256, 0, stream>>>(x, norm1_w, norm1_b, xn_bf, xf);
  lambda_kernel<<<1, 128, 0, stream>>>(lambda_q, lam);

  const int MT16 = MTOK / 16;   // 784 (single-tile M tiles)
  const int MT32 = MTOK / 32;   // 392 (multi-tile M tiles)
  for (int m = 0; m < 2; ++m) {
    // xz = xn @ in_proj^T  (M x 768): 392*12 = 4704 waves -> 588 blocks
    gemm_bf16_mt_kernel<0, 384><<<(MT32 * 12 + 7) / 8, 256, 0, stream>>>(
        xn_bf, wb[m][0], nullptr, xz, MTOK, 768);
    // depthwise causal conv + SiLU
    conv_silu_kernel<<<(MTOK * DIMC) / 256, 256, 0, stream>>>(xz, mw[m][1], mw[m][2], xs, xsb);
    // x_dbl = xs @ x_proj^T  (M x 48, cols 40..47 from zero-padded weights)
    gemm_bf16_kernel<<<(MT16 * 3 + 7) / 8, 256, 0, stream>>>(xsb, wb[m][1], xdbl, MTOK, 48, 384);
    // dt input pack (M x 32 bf16)
    pack_dt_kernel<<<(MTOK * 32) / 256, 256, 0, stream>>>(xdbl, dtin);
    // dt = softplus(dtin @ dt_proj^T + b)  (M x 384): 392*6 waves -> 294 blocks
    gemm_bf16_mt_kernel<1, 32><<<(MT32 * 6 + 7) / 8, 256, 0, stream>>>(
        dtin, wb[m][2], mw[m][5], dtb, MTOK, 384);
    // chunked selective scan
    scan_phase1_kernel<<<(NB * DIMC * NCHUNK) / 256, 256, 0, stream>>>(dtb, xs, xdbl, mw[m][6], cs);
    scan_fix_kernel<<<(NB * DIMC * NSTATE) / 256, 256, 0, stream>>>(cs, hin);
    scan_phase3_kernel<<<(NB * DIMC * NCHUNK) / 256, 256, 0, stream>>>(dtb, xs, xdbl, mw[m][6], mw[m][7], xz, hin, gb);
    // y_m = g @ out_proj^T  (M x 384)
    gemm_bf16_mt_kernel<0, 384><<<(MT32 * 6 + 7) / 8, 256, 0, stream>>>(
        gb, wb[m][3], nullptr, ybuf[m], MTOK, 384);
  }

  // attn = LN(y1 - lam*y2); out = LN(xf + attn); transpose to (B,C,L).
  final_kernel<<<NB * (LSEQ / 32), 256, 0, stream>>>(xf, ybuf0, ybuf1, lam,
                                                     subln_w, subln_b, norm2_w, norm2_b,
                                                     (float*)d_out);
}
